// TFMambaModel_53738630808038
// MI455X (gfx1250) — matrix operand order repaired and checked by reading, not compiled, not run on hardware
//
#include <hip/hip_runtime.h>
#include <hip/hip_bf16.h>

typedef __attribute__((ext_vector_type(16))) __bf16 v16bf;
typedef __attribute__((ext_vector_type(8)))  float  v8f;

#define NB 8
#define HDIM 128
#define DIN 256

// ---------------------------------------------------------------------------
// rfft magnitude: Xf[b,k] = | sum_n x[b,n] e^{-2pi i k n / 4096} |, k=0..2048
// Exact integer phase reduction keeps sincos argument in [-2pi, 0].
// ---------------------------------------------------------------------------
__global__ __launch_bounds__(256) void dft_abs_kernel(const float* __restrict__ x,
                                                      float* __restrict__ xf) {
    int gid = blockIdx.x * 256 + threadIdx.x;
    int b = gid / 2049, k = gid % 2049;
    if (b >= NB) return;
    const float* xb = x + b * 4096;
    const float w = -6.283185307179586f / 4096.0f;
    float re = 0.f, im = 0.f;
    for (int n = 0; n < 4096; ++n) {
        int m = (k * n) & 4095;
        float s, c;
        __sincosf(w * (float)m, &s, &c);
        float v = xb[n];
        re = fmaf(v, c, re);
        im = fmaf(v, s, im);
    }
    xf[b * 2049 + k] = sqrtf(re * re + im * im);
}

// ---------------------------------------------------------------------------
// Embed conv1d: in_ch=1, out_ch=128, k=16, stride=4, pad (6,6). Writes (B,L,128).
// ---------------------------------------------------------------------------
__global__ __launch_bounds__(256) void embed_conv_kernel(const float* __restrict__ in,
                                                         int Lin, int Lout,
                                                         const float* __restrict__ w,
                                                         const float* __restrict__ bias,
                                                         float* __restrict__ out) {
    int gid = blockIdx.x * 256 + threadIdx.x;
    int c = gid & 127;
    int l = (gid >> 7) % Lout;
    int b = gid / (128 * Lout);
    if (b >= NB) return;
    const float* ib = in + (size_t)b * Lin;
    float acc = bias[c];
    int base = l * 4 - 6;
#pragma unroll
    for (int j = 0; j < 16; ++j) {
        int idx = base + j;
        if (idx >= 0 && idx < Lin) acc = fmaf(ib[idx], w[c * 16 + j], acc);
    }
    out[((size_t)b * Lout + l) * HDIM + c] = acc;
}

// ---------------------------------------------------------------------------
// GroupNorm(groups=8) + affine + SiLU, in-place on (B,L,128). Block per (b,g).
// ---------------------------------------------------------------------------
__global__ __launch_bounds__(256) void gn_silu_kernel(float* __restrict__ E, int L,
                                                      const float* __restrict__ g,
                                                      const float* __restrict__ bta) {
    int b = blockIdx.x >> 3;
    int gr = blockIdx.x & 7;
    int cnt = L * 16;
    float s = 0.f, ss = 0.f;
    for (int i = threadIdx.x; i < cnt; i += 256) {
        int l = i >> 4, c = gr * 16 + (i & 15);
        float v = E[((size_t)b * L + l) * HDIM + c];
        s += v; ss += v * v;
    }
    __shared__ float rs[256], rq[256];
    rs[threadIdx.x] = s; rq[threadIdx.x] = ss;
    __syncthreads();
    for (int off = 128; off > 0; off >>= 1) {
        if (threadIdx.x < off) { rs[threadIdx.x] += rs[threadIdx.x + off];
                                 rq[threadIdx.x] += rq[threadIdx.x + off]; }
        __syncthreads();
    }
    float mu  = rs[0] / (float)cnt;
    float var = rq[0] / (float)cnt - mu * mu;
    float inv = rsqrtf(var + 1e-5f);
    for (int i = threadIdx.x; i < cnt; i += 256) {
        int l = i >> 4, c = gr * 16 + (i & 15);
        size_t idx = ((size_t)b * L + l) * HDIM + c;
        float v = (E[idx] - mu) * inv * g[c] + bta[c];
        E[idx] = v / (1.f + __expf(-v));   // silu
    }
}

// ---------------------------------------------------------------------------
// LayerNorm over last dim (128). One 128-thread block per row.
// ---------------------------------------------------------------------------
__global__ __launch_bounds__(128) void ln_kernel(const float* __restrict__ X,
                                                 float* __restrict__ H,
                                                 const float* __restrict__ g,
                                                 const float* __restrict__ bta) {
    int r = blockIdx.x;
    int c = threadIdx.x;
    float v = X[(size_t)r * HDIM + c];
    __shared__ float rs[128], rq[128];
    rs[c] = v; rq[c] = v * v;
    __syncthreads();
    for (int off = 64; off > 0; off >>= 1) {
        if (c < off) { rs[c] += rs[c + off]; rq[c] += rq[c + off]; }
        __syncthreads();
    }
    float mu  = rs[0] * (1.f / 128.f);
    float var = rq[0] * (1.f / 128.f) - mu * mu;
    H[(size_t)r * HDIM + c] = (v - mu) * rsqrtf(var + 1e-5f) * g[c] + bta[c];
}

// ---------------------------------------------------------------------------
// WMMA bf16 GEMM: C[M,N] = A[M,K] * B[N,K]^T   (fp32 in/out, bf16 matrix pipe)
// Block: 256 thr = 8 waves; block tile 128(M) x 64(N); wave tile 16(M) x 64(N):
// one A fragment feeds 4 back-to-back v_wmma_f32_16x16x32_bf16 per K-step.
// Staging is split load-phase / store-phase so all 12 global_load_b64 per
// K-step issue as one clause covered by a single s_wait_loadcnt.
// Requires M%128==0, N%64==0, K%32==0 (true for all call sites).
// ---------------------------------------------------------------------------
__global__ __launch_bounds__(256) void gemm_bf16_wmma(const float* __restrict__ A,
                                                      const float* __restrict__ B,
                                                      float* __restrict__ C,
                                                      int M, int N, int K) {
    __shared__ __bf16 As[128][32];  // 8 KB
    __shared__ __bf16 Bs[64][32];   // 4 KB
    const int t    = threadIdx.x;
    const int wave = t >> 5;
    const int lane = t & 31;
    const int m0 = blockIdx.x * 128;
    const int n0 = blockIdx.y * 64;
    const int half = lane >> 4;
    const int mr   = lane & 15;

    // fixed per-thread staging coordinates
    int arow[8], akp[8];
#pragma unroll
    for (int j = 0; j < 8; ++j) { int idx = j * 256 + t; arow[j] = idx >> 4; akp[j] = idx & 15; }
    int brow[4], bkp[4];
#pragma unroll
    for (int j = 0; j < 4; ++j) { int idx = j * 256 + t; brow[j] = idx >> 4; bkp[j] = idx & 15; }

    v8f acc[4] = {{}, {}, {}, {}};
    for (int k0 = 0; k0 < K; k0 += 32) {
        // phase 1: issue all global loads (independent -> single clause/wait)
        float2 areg[8], breg[4];
#pragma unroll
        for (int j = 0; j < 8; ++j)
            areg[j] = *(const float2*)(A + (size_t)(m0 + arow[j]) * K + k0 + akp[j] * 2);
#pragma unroll
        for (int j = 0; j < 4; ++j)
            breg[j] = *(const float2*)(B + (size_t)(n0 + brow[j]) * K + k0 + bkp[j] * 2);
        // L2->WGP prefetch of next K-step's A tile (global_prefetch_b8)
        if (k0 + 32 < K)
            __builtin_prefetch(A + (size_t)(m0 + (t >> 1)) * K + k0 + 32 + (t & 1) * 16, 0, 3);
        // phase 2: convert + LDS store
#pragma unroll
        for (int j = 0; j < 8; ++j) {
            As[arow[j]][akp[j] * 2]     = (__bf16)areg[j].x;
            As[arow[j]][akp[j] * 2 + 1] = (__bf16)areg[j].y;
        }
#pragma unroll
        for (int j = 0; j < 4; ++j) {
            Bs[brow[j]][bkp[j] * 2]     = (__bf16)breg[j].x;
            Bs[brow[j]][bkp[j] * 2 + 1] = (__bf16)breg[j].y;
        }
        __syncthreads();

        // A fragment, contiguous k = 16*half + s (same as B): row = wave*16 + mr, K-pairs split by half
        union { unsigned u[8]; v16bf v; } afrag;
        const unsigned* Ar = (const unsigned*)&As[wave * 16 + mr][0]; // 16 dwords/row
#pragma unroll
        for (int v = 0; v < 8; ++v) {
            int ka = half * 8 + v;
            afrag.u[v] = Ar[ka];
        }
        // 4 N-subtiles against the same A fragment
#pragma unroll
        for (int nt = 0; nt < 4; ++nt) {
            union { unsigned u[8]; v16bf v; } bfrag;
            const unsigned* Br = (const unsigned*)&Bs[nt * 16 + mr][0];
#pragma unroll
            for (int v = 0; v < 8; ++v) bfrag.u[v] = Br[half * 8 + v]; // K=16*half+2v
            acc[nt] = __builtin_amdgcn_wmma_f32_16x16x32_bf16(
                false, afrag.v, false, bfrag.v, (short)0, acc[nt], false, false);
        }
        __syncthreads();
    }
    // C/D layout: VGPR v -> row (v + 8*half), col (lane&15)
#pragma unroll
    for (int nt = 0; nt < 4; ++nt) {
#pragma unroll
        for (int v = 0; v < 8; ++v) {
            int m = m0 + wave * 16 + v + 8 * half;
            C[(size_t)m * N + n0 + nt * 16 + mr] = acc[nt][v];
        }
    }
}

// ---------------------------------------------------------------------------
// Depthwise causal conv (K=4, left pad 3) on xc = XZ[...,:256], + bias, SiLU.
// ---------------------------------------------------------------------------
__global__ __launch_bounds__(256) void mamba_conv_kernel(const float* __restrict__ XZ, int L,
                                                         const float* __restrict__ w,
                                                         const float* __restrict__ bias,
                                                         float* __restrict__ XC) {
    int gid = blockIdx.x * 256 + threadIdx.x;
    int d = gid & 255;
    int l = (gid >> 8) % L;
    int b = gid / (256 * L);
    if (b >= NB) return;
    float acc = bias[d];
#pragma unroll
    for (int j = 0; j < 4; ++j) {
        int ls = l - 3 + j;
        if (ls >= 0)
            acc = fmaf(XZ[((size_t)b * L + ls) * 512 + d], w[d * 4 + j], acc);
    }
    XC[((size_t)b * L + l) * DIN + d] = acc / (1.f + __expf(-acc)); // silu
}

// dbl[m,e] = dot(XC[m,:256], xproj_w[e,:]),  e < 40
__global__ __launch_bounds__(256) void xproj_kernel(const float* __restrict__ XC,
                                                    const float* __restrict__ w,
                                                    float* __restrict__ DBL, int Mrows) {
    int gid = blockIdx.x * 256 + threadIdx.x;
    int e = gid % 40;
    int m = gid / 40;
    if (m >= Mrows) return;
    const float* xr = XC + (size_t)m * DIN;
    const float* wr = w + e * DIN;
    float acc = 0.f;
    for (int d = 0; d < DIN; ++d) acc = fmaf(xr[d], wr[d], acc);
    DBL[(size_t)m * 40 + e] = acc;
}

// dt[m,d] = softplus(dot(dbl[m,:8], dt_w[d,:]) + dt_b[d])
__global__ __launch_bounds__(256) void dt_kernel(const float* __restrict__ DBL,
                                                 const float* __restrict__ w,
                                                 const float* __restrict__ bias,
                                                 float* __restrict__ DT, int Mrows) {
    int gid = blockIdx.x * 256 + threadIdx.x;
    int d = gid & 255;
    int m = gid >> 8;
    if (m >= Mrows) return;
    const float* r = DBL + (size_t)m * 40;
    float acc = bias[d];
#pragma unroll
    for (int j = 0; j < 8; ++j) acc = fmaf(r[j], w[d * 8 + j], acc);
    DT[(size_t)m * DIN + d] = (acc > 20.f) ? acc : log1pf(__expf(acc));
}

// ---------------------------------------------------------------------------
// Selective scan: one thread per (b,d); 16-element state + A in registers.
// Y[b,l,d] = (sum_n h*C + xc*D) * silu(z),  h = exp(dt*A)*h + dt*xc*B.
// ---------------------------------------------------------------------------
__global__ __launch_bounds__(256) void scan_kernel(const float* __restrict__ DT,
                                                   const float* __restrict__ XC,
                                                   const float* __restrict__ XZ,
                                                   const float* __restrict__ DBL,
                                                   const float* __restrict__ Alog,
                                                   const float* __restrict__ Dp,
                                                   float* __restrict__ Y, int L) {
    int b = blockIdx.x;
    int d = threadIdx.x;
    float A[16], h[16];
#pragma unroll
    for (int n = 0; n < 16; ++n) { A[n] = -__expf(Alog[d * 16 + n]); h[n] = 0.f; }
    float Dd = Dp[d];
    for (int l = 0; l < L; ++l) {
        size_t m = (size_t)b * L + l;
        float dt  = DT[m * DIN + d];
        float xcv = XC[m * DIN + d];
        float z   = XZ[m * 512 + DIN + d];
        const float* bc = DBL + m * 40 + 8;   // B: [0..15], C: [16..31] (broadcast)
        float dtx = dt * xcv;
        float acc = 0.f;
#pragma unroll
        for (int n = 0; n < 16; ++n) {
            h[n] = __expf(dt * A[n]) * h[n] + dtx * bc[n];
            acc  = fmaf(h[n], bc[16 + n], acc);
        }
        float yv = acc + xcv * Dd;
        Y[m * DIN + d] = yv * (z / (1.f + __expf(-z)));
    }
}

__global__ __launch_bounds__(256) void flip_kernel(const float* __restrict__ src,
                                                   float* __restrict__ dst, int L) {
    int gid = blockIdx.x * 256 + threadIdx.x;
    int c = gid & 127;
    int l = (gid >> 7) % L;
    int b = gid / (128 * L);
    if (b >= NB) return;
    dst[((size_t)b * L + l) * HDIM + c] = src[((size_t)b * L + (L - 1 - l)) * HDIM + c];
}

__global__ __launch_bounds__(256) void add_kernel(float* __restrict__ X,
                                                  const float* __restrict__ O,
                                                  int L, int flip) {
    int gid = blockIdx.x * 256 + threadIdx.x;
    int c = gid & 127;
    int l = (gid >> 7) % L;
    int b = gid / (128 * L);
    if (b >= NB) return;
    int ls = flip ? (L - 1 - l) : l;
    X[((size_t)b * L + l) * HDIM + c] += O[((size_t)b * L + ls) * HDIM + c];
}

__global__ __launch_bounds__(256) void meanpool_kernel(const float* __restrict__ ET,
                                                       const float* __restrict__ EF,
                                                       float* __restrict__ feat) {
    int gid = blockIdx.x * 256 + threadIdx.x;
    int c = gid & 255, b = gid >> 8;
    if (b >= NB) return;
    float s = 0.f;
    if (c < 128) {
        for (int l = 0; l < 1024; ++l) s += ET[((size_t)b * 1024 + l) * HDIM + c];
        s *= (1.f / 1024.f);
    } else {
        int cc = c - 128;
        for (int l = 0; l < 512; ++l) s += EF[((size_t)b * 512 + l) * HDIM + cc];
        s *= (1.f / 512.f);
    }
    feat[b * 256 + c] = s;
}

// h1 linear -> GN(8 groups over 128x1) + silu -> h2 linear. Block per batch.
__global__ __launch_bounds__(128) void head_kernel(const float* __restrict__ feat,
                                                   const float* __restrict__ h1w,
                                                   const float* __restrict__ h1b,
                                                   const float* __restrict__ gg,
                                                   const float* __restrict__ gb,
                                                   const float* __restrict__ h2w,
                                                   const float* __restrict__ h2b,
                                                   float* __restrict__ out) {
    int b = blockIdx.x;
    int o = threadIdx.x;
    const float* fr = feat + b * 256;
    float acc = h1b[o];
    for (int j = 0; j < 256; ++j) acc = fmaf(fr[j], h1w[o * 256 + j], acc);
    __shared__ float y[128], sact[128];
    y[o] = acc;
    __syncthreads();
    int g = o >> 4;
    float mu = 0.f, ss = 0.f;
    for (int j = 0; j < 16; ++j) { float v = y[g * 16 + j]; mu += v; ss += v * v; }
    mu *= (1.f / 16.f);
    ss = ss * (1.f / 16.f) - mu * mu;
    float v = (y[o] - mu) * rsqrtf(ss + 1e-5f) * gg[o] + gb[o];
    sact[o] = v / (1.f + __expf(-v));
    __syncthreads();
    if (o < 10) {
        float a2 = h2b[o];
        for (int j = 0; j < 128; ++j) a2 = fmaf(sact[j], h2w[o * 128 + j], a2);
        out[b * 10 + o] = a2;
    }
}

// ---------------------------------------------------------------------------
static void run_mamba(const float* H, int L, int layer, void* const* d_in,
                      float* XZ, float* XC, float* DBL, float* DT, float* Y,
                      float* OUT, hipStream_t stream) {
    const float* in_w    = (const float*)d_in[11] + (size_t)layer * 512 * 128;
    const float* conv_w  = (const float*)d_in[12] + (size_t)layer * 256 * 4;
    const float* conv_b  = (const float*)d_in[13] + (size_t)layer * 256;
    const float* xproj_w = (const float*)d_in[14] + (size_t)layer * 40 * 256;
    const float* dt_w    = (const float*)d_in[15] + (size_t)layer * 256 * 8;
    const float* dt_b    = (const float*)d_in[16] + (size_t)layer * 256;
    const float* Alog    = (const float*)d_in[17] + (size_t)layer * 256 * 16;
    const float* Dpar    = (const float*)d_in[18] + (size_t)layer * 256;
    const float* out_w   = (const float*)d_in[19] + (size_t)layer * 128 * 256;
    int M = NB * L;
    gemm_bf16_wmma<<<dim3(M / 128, 512 / 64), 256, 0, stream>>>(H, in_w, XZ, M, 512, 128);
    mamba_conv_kernel<<<(NB * L * 256) / 256, 256, 0, stream>>>(XZ, L, conv_w, conv_b, XC);
    xproj_kernel<<<(M * 40 + 255) / 256, 256, 0, stream>>>(XC, xproj_w, DBL, M);
    dt_kernel<<<(M * 256) / 256, 256, 0, stream>>>(DBL, dt_w, dt_b, DT, M);
    scan_kernel<<<NB, 256, 0, stream>>>(DT, XC, XZ, DBL, Alog, Dpar, Y, L);
    gemm_bf16_wmma<<<dim3(M / 128, 128 / 64), 256, 0, stream>>>(Y, out_w, OUT, M, 128, 256);
}

extern "C" void kernel_launch(void* const* d_in, const int* in_sizes, int n_in,
                              void* d_out, int out_size, void* d_ws, size_t ws_size,
                              hipStream_t stream) {
    (void)in_sizes; (void)n_in; (void)out_size; (void)ws_size;
    const float* x       = (const float*)d_in[0];
    const float* t_cw    = (const float*)d_in[1];
    const float* t_cb    = (const float*)d_in[2];
    const float* t_gg    = (const float*)d_in[3];
    const float* t_gb    = (const float*)d_in[4];
    const float* f_cw    = (const float*)d_in[5];
    const float* f_cb    = (const float*)d_in[6];
    const float* f_gg    = (const float*)d_in[7];
    const float* f_gb    = (const float*)d_in[8];
    const float* ln_g    = (const float*)d_in[9];
    const float* ln_b    = (const float*)d_in[10];
    const float* h1_w    = (const float*)d_in[20];
    const float* h1_b    = (const float*)d_in[21];
    const float* h_gn_g  = (const float*)d_in[22];
    const float* h_gn_b  = (const float*)d_in[23];
    const float* h2_w    = (const float*)d_in[24];
    const float* h2_b    = (const float*)d_in[25];

    float* ws = (float*)d_ws;
    size_t off = 0;
    auto carve = [&](size_t n) { float* p = ws + off; off += n; return p; };
    float* XF   = carve(16512);             // 8 x 2049
    float* ET   = carve(8 * 1024 * 128);    // x_t residual
    float* EF   = carve(8 * 512 * 128);     // x_f residual
    float* H    = carve(8 * 1024 * 128);    // LN output
    float* HR   = carve(8 * 512 * 128);     // flipped LN output
    float* XZ   = carve(8 * 1024 * 512);
    float* XC   = carve(8 * 1024 * 256);
    float* DBL  = carve(8 * 1024 * 40);
    float* DT   = carve(8 * 1024 * 256);
    float* Y    = carve(8 * 1024 * 256);
    float* O1   = carve(8 * 1024 * 128);
    float* O2   = carve(8 * 512 * 128);
    float* FEAT = carve(8 * 256);

    // frequency branch input + both embeds
    dft_abs_kernel<<<(8 * 2049 + 255) / 256, 256, 0, stream>>>(x, XF);
    embed_conv_kernel<<<(8 * 1024 * 128) / 256, 256, 0, stream>>>(x, 4096, 1024, t_cw, t_cb, ET);
    gn_silu_kernel<<<64, 256, 0, stream>>>(ET, 1024, t_gg, t_gb);
    embed_conv_kernel<<<(8 * 512 * 128) / 256, 256, 0, stream>>>(XF, 2049, 512, f_cw, f_cb, EF);
    gn_silu_kernel<<<64, 256, 0, stream>>>(EF, 512, f_gg, f_gb);

    // time-path Mamba layers (L=1024)
    for (int l = 0; l < 2; ++l) {
        ln_kernel<<<8 * 1024, 128, 0, stream>>>(ET, H, ln_g + 128 * l, ln_b + 128 * l);
        run_mamba(H, 1024, l, d_in, XZ, XC, DBL, DT, Y, O1, stream);
        add_kernel<<<(8 * 1024 * 128) / 256, 256, 0, stream>>>(ET, O1, 1024, 0);
    }
    // frequency-path bidirectional Mamba layers (L=512)
    for (int l = 0; l < 2; ++l) {
        ln_kernel<<<8 * 512, 128, 0, stream>>>(EF, H, ln_g + 128 * (2 + l), ln_b + 128 * (2 + l));
        run_mamba(H, 512, 2 + 2 * l, d_in, XZ, XC, DBL, DT, Y, O1, stream);
        flip_kernel<<<(8 * 512 * 128) / 256, 256, 0, stream>>>(H, HR, 512);
        run_mamba(HR, 512, 3 + 2 * l, d_in, XZ, XC, DBL, DT, Y, O2, stream);
        add_kernel<<<(8 * 512 * 128) / 256, 256, 0, stream>>>(EF, O1, 512, 0);
        add_kernel<<<(8 * 512 * 128) / 256, 256, 0, stream>>>(EF, O2, 512, 1);
    }

    meanpool_kernel<<<8, 256, 0, stream>>>(ET, EF, FEAT);
    head_kernel<<<8, 128, 0, stream>>>(FEAT, h1_w, h1_b, h_gn_g, h_gn_b, h2_w, h2_b,
                                       (float*)d_out);
}
